// GRU_76501957476752
// MI455X (gfx1250) — matrix operand order repaired and checked by reading, not compiled, not run on hardware
//
#include <hip/hip_runtime.h>
#include <hip/hip_bf16.h>
#include <stdint.h>
#include <stddef.h>

// Problem dims
#define BB 64
#define SS 5
#define HH 50
#define TT 30
#define DD 300
#define FF 400
#define QQ 200

#define HSTRIDE 416      // F padded to K-multiple of 32 for bf16 WMMA
#define PSTRIDE 208      // Q padded to 16
#define IN_ELEMS 10240   // 34 padded rows * 300 + slack for K overrun
#define CONV_KC 29       // ceil(900/32)
#define CONV_NT 25       // 400/16
#define ATT_KC 13        // 416/32
#define ATT_NT 13        // 208/16
#define GRU_KC 13        // 416/32
#define GRU_NT 75        // 1200/16

typedef __attribute__((ext_vector_type(16))) __bf16 v16bf;
typedef __attribute__((ext_vector_type(8)))  float  v8f;
typedef __attribute__((ext_vector_type(2)))  int    i2v;
typedef __attribute__((ext_vector_type(4)))  int    i4v;

union FragBF {
    v16bf v;
    i2v   d2[4];
    i4v   d4[2];
};

// ---------------------------------------------------------------------------
// Pack a K x N (fp32) matrix into bf16 B-fragment order for wmma 16x16x32:
// layout: [nt][kc][lane][16 elems]; lane: n = lane&15, klocal = 16*(j>>3)+8*(lane>>4)+(j&7)
// ---------------------------------------------------------------------------
__global__ void pack_b_kernel(const float* __restrict__ src, __bf16* __restrict__ dst,
                              int Kreal, int Nreal, int KC, int NT, int src_ld,
                              int transpose)
{
    int idx = blockIdx.x * blockDim.x + threadIdx.x;
    int total = NT * KC * 512;
    if (idx >= total) return;
    int j     = idx & 15;
    int lane  = (idx >> 4) & 31;
    int chunk = idx >> 9;
    int kc    = chunk % KC;
    int nt    = chunk / KC;
    int n      = lane & 15;
    int klocal = ((j >> 3) << 4) + ((lane >> 4) << 3) + (j & 7);
    int k      = kc * 32 + klocal;
    int col    = nt * 16 + n;
    float v = 0.0f;
    if (k < Kreal && col < Nreal)
        v = transpose ? src[(size_t)col * src_ld + k]
                      : src[(size_t)k * src_ld + col];
    dst[idx] = (__bf16)v;
}

// ---------------------------------------------------------------------------
// Fused title encoder: conv1d(SAME,W=3)+ReLU -> additive attention -> LayerNorm
// One block (8 waves) per title. Each wave owns N-tiles and blocks both M-tiles
// so every B fragment is loaded once and used by two WMMAs.
// ---------------------------------------------------------------------------
__global__ void __launch_bounds__(256)
encoder_kernel(const float* __restrict__ x,        // [N, T, D]
               const __bf16* __restrict__ convB,   // packed [25][29][32][16]
               const float* __restrict__ conv_b,   // [400]
               const __bf16* __restrict__ attWB,   // packed [13][13][32][16]
               const float* __restrict__ att_b,    // [200]
               const float* __restrict__ att_q,    // [200]
               const float* __restrict__ ln_g,     // [400]
               const float* __restrict__ ln_b,     // [400]
               float*  __restrict__ out_f32,       // [N, 400]
               __bf16* __restrict__ out_bf16)      // [N, 416]
{
    __shared__ __align__(16) unsigned char smem0[26624]; // input (bf16) / P (f32) / rep+reduce
    __shared__ __align__(16) unsigned char smem1[26624]; // h (bf16, 32 x 416)
    __shared__ float s_bias[FF];
    __shared__ float s_ab[QQ];
    __shared__ float s_aq[QQ];
    __shared__ float s_sp[32 * 8];
    __shared__ float s_sc[32];
    __shared__ float s_a[32];
    __shared__ float s_stat[2];

    const int tid  = threadIdx.x;
    const int lane = tid & 31;
    const int wave = tid >> 5;
    const int n    = blockIdx.x;

    __bf16* lds_in = (__bf16*)smem0;
    __bf16* lds_h  = (__bf16*)smem1;

    // ---- phase 0: zero + stage input (one zero pad row each side, stride 300
    //      so im2col K index k = w*300+d is contiguous: A(t,k) = lds_in[t*300+k])
    for (int i = tid; i < IN_ELEMS; i += 256)     lds_in[i] = (__bf16)0.0f;
    for (int i = tid; i < 32 * HSTRIDE; i += 256) lds_h[i]  = (__bf16)0.0f;
    for (int i = tid; i < FF; i += 256)           s_bias[i] = conv_b[i];
    for (int i = tid; i < QQ; i += 256) { s_ab[i] = att_b[i]; s_aq[i] = att_q[i]; }
    __syncthreads();
    const float* xb = x + (size_t)n * (TT * DD);
    for (int i = tid; i < TT * DD; i += 256)
        lds_in[DD + i] = (__bf16)xb[i];
    __syncthreads();

    const int m     = lane & 15;
    const int khalf = (lane >> 4) << 3;
    const int mbase = (lane >> 4) << 3;

    // ---- phase 1: conv as GEMM  h[32x400] = A[32x928] * B[928x400], +bias, ReLU
    for (int nt = wave; nt < CONV_NT; nt += 8) {
        v8f acc0 = {}, acc1 = {};
        for (int kc = 0; kc < CONV_KC; ++kc) {
            FragBF a0, a1, b;
            const __bf16* bp = convB + (((size_t)nt * CONV_KC + kc) * 32 + lane) * 16;
            b.d4[0] = *(const i4v*)bp;
            b.d4[1] = *(const i4v*)(bp + 8);
            int base0 = m * DD + kc * 32 + khalf;          // mult of 4 -> 8B aligned
            a0.d2[0] = *(const i2v*)&lds_in[base0];
            a0.d2[1] = *(const i2v*)&lds_in[base0 + 4];
            a0.d2[2] = *(const i2v*)&lds_in[base0 + 16];
            a0.d2[3] = *(const i2v*)&lds_in[base0 + 20];
            int base1 = base0 + 16 * DD;
            a1.d2[0] = *(const i2v*)&lds_in[base1];
            a1.d2[1] = *(const i2v*)&lds_in[base1 + 4];
            a1.d2[2] = *(const i2v*)&lds_in[base1 + 16];
            a1.d2[3] = *(const i2v*)&lds_in[base1 + 20];
            acc0 = __builtin_amdgcn_wmma_f32_16x16x32_bf16(
                       false, a0.v, false, b.v, (short)0, acc0, false, false);
            acc1 = __builtin_amdgcn_wmma_f32_16x16x32_bf16(
                       false, a1.v, false, b.v, (short)0, acc1, false, false);
        }
        int col   = nt * 16 + m;
        float bia = s_bias[col];
        for (int v = 0; v < 8; ++v) {
            float h0 = acc0[v] + bia;
            float h1 = acc1[v] + bia;
            lds_h[(mbase + v) * HSTRIDE + col]        = (__bf16)(h0 > 0.0f ? h0 : 0.0f);
            lds_h[(16 + mbase + v) * HSTRIDE + col]   = (__bf16)(h1 > 0.0f ? h1 : 0.0f);
        }
    }
    __syncthreads();

    // ---- phase 2: P[32x208] = h[32x416] * attW[416x208]
    float* lds_P = (float*)smem0;
    for (int nt = wave; nt < ATT_NT; nt += 8) {
        v8f acc0 = {}, acc1 = {};
        for (int kc = 0; kc < ATT_KC; ++kc) {
            FragBF a0, a1, b;
            const __bf16* bp = attWB + (((size_t)nt * ATT_KC + kc) * 32 + lane) * 16;
            b.d4[0] = *(const i4v*)bp;
            b.d4[1] = *(const i4v*)(bp + 8);
            int base0 = m * HSTRIDE + kc * 32 + khalf;     // mult of 8 -> 16B aligned
            a0.d4[0] = *(const i4v*)&lds_h[base0];
            a0.d4[1] = *(const i4v*)&lds_h[base0 + 16];
            int base1 = base0 + 16 * HSTRIDE;
            a1.d4[0] = *(const i4v*)&lds_h[base1];
            a1.d4[1] = *(const i4v*)&lds_h[base1 + 16];
            acc0 = __builtin_amdgcn_wmma_f32_16x16x32_bf16(
                       false, a0.v, false, b.v, (short)0, acc0, false, false);
            acc1 = __builtin_amdgcn_wmma_f32_16x16x32_bf16(
                       false, a1.v, false, b.v, (short)0, acc1, false, false);
        }
        int col = nt * 16 + m;
        for (int v = 0; v < 8; ++v) {
            lds_P[(mbase + v) * PSTRIDE + col]      = acc0[v];
            lds_P[(16 + mbase + v) * PSTRIDE + col] = acc1[v];
        }
    }
    __syncthreads();

    // ---- phase 3: scores s[t] = sum_q tanh(P+b)*q ; softmax over t<30
    {
        int t = tid >> 3, j = tid & 7;
        float p = 0.0f;
        for (int q = j * 25; q < j * 25 + 25; ++q)
            p += tanhf(lds_P[t * PSTRIDE + q] + s_ab[q]) * s_aq[q];
        s_sp[t * 8 + j] = p;
    }
    __syncthreads();
    if (tid < 32) {
        float s = 0.0f;
        for (int j = 0; j < 8; ++j) s += s_sp[tid * 8 + j];
        s_sc[tid] = s;
    }
    __syncthreads();
    if (tid == 0) {
        float mx = s_sc[0];
        for (int t = 1; t < TT; ++t) mx = fmaxf(mx, s_sc[t]);
        float den = 0.0f;
        for (int t = 0; t < TT; ++t) { float e = expf(s_sc[t] - mx); s_a[t] = e; den += e; }
        float inv = 1.0f / den;
        for (int t = 0; t < TT; ++t) s_a[t] *= inv;
        s_a[30] = 0.0f; s_a[31] = 0.0f;
    }
    __syncthreads();

    // ---- phase 4: rep = sum_t a[t] h[t,:]; LayerNorm; write f32 + bf16
    float* lds_rep = (float*)smem0;               // P no longer needed
    float* red1    = (float*)(smem0 + 2048);
    float* red2    = (float*)(smem0 + 3072);
    for (int f = tid; f < FF; f += 256) {
        float r = 0.0f;
        for (int t = 0; t < TT; ++t)
            r += s_a[t] * (float)lds_h[t * HSTRIDE + f];
        lds_rep[f] = r;
    }
    __syncthreads();
    {
        float a0 = lds_rep[tid];
        float a1 = (tid + 256 < FF) ? lds_rep[tid + 256] : 0.0f;
        red1[tid] = a0 + a1;
        red2[tid] = a0 * a0 + a1 * a1;
    }
    __syncthreads();
    for (int off = 128; off > 0; off >>= 1) {
        if (tid < off) { red1[tid] += red1[tid + off]; red2[tid] += red2[tid + off]; }
        __syncthreads();
    }
    if (tid == 0) {
        float mean = red1[0] * (1.0f / FF);
        float var  = red2[0] * (1.0f / FF) - mean * mean;
        s_stat[0] = mean;
        s_stat[1] = rsqrtf(var + 1e-5f);
    }
    __syncthreads();
    float mean = s_stat[0], inv = s_stat[1];
    for (int f = tid; f < FF; f += 256) {
        float val = (lds_rep[f] - mean) * inv * ln_g[f] + ln_b[f];
        out_f32[(size_t)n * FF + f]       = val;
        out_bf16[(size_t)n * HSTRIDE + f] = (__bf16)val;
    }
    if (tid < HSTRIDE - FF)
        out_bf16[(size_t)n * HSTRIDE + FF + tid] = (__bf16)0.0f;
}

// ---------------------------------------------------------------------------
// GRU step GEMMs: gi = x_t * Wih^T, gh = h * Whh^T   (M=64, N=1200, K=416pad)
// blockIdx.y selects gi (0) / gh (1). Each wave owns one N-tile and all 4
// M-tiles: each B fragment is loaded once and feeds 4 WMMAs.
// ---------------------------------------------------------------------------
__global__ void __launch_bounds__(256)
gru_gemm_kernel(const __bf16* __restrict__ x_bf16,  // [B*H, 416]
                const __bf16* __restrict__ h_bf16,  // [B, 416]
                const __bf16* __restrict__ wihB,    // packed [75][13][32][16]
                const __bf16* __restrict__ whhB,
                float* __restrict__ gi,             // [64, 1200]
                float* __restrict__ gh,
                int t)
{
    const int lane = threadIdx.x & 31;
    const int wave = threadIdx.x >> 5;
    int nt = blockIdx.x * 8 + wave;
    if (nt >= GRU_NT) return;

    const __bf16* A; int rstride; const __bf16* Bp; float* out;
    if (blockIdx.y == 0) { A = x_bf16 + (size_t)t * HSTRIDE; rstride = HH * HSTRIDE; Bp = wihB; out = gi; }
    else                 { A = h_bf16;                       rstride = HSTRIDE;      Bp = whhB; out = gh; }

    const int m     = lane & 15;
    const int khalf = (lane >> 4) << 3;
    v8f acc[4] = {};
    for (int kc = 0; kc < GRU_KC; ++kc) {
        FragBF b;
        const __bf16* bp = Bp + (((size_t)nt * GRU_KC + kc) * 32 + lane) * 16;
        b.d4[0] = *(const i4v*)bp;
        b.d4[1] = *(const i4v*)(bp + 8);
#pragma unroll
        for (int mt = 0; mt < 4; ++mt) {
            FragBF a;
            const __bf16* ap = A + (size_t)(mt * 16 + m) * rstride + kc * 32 + khalf;
            a.d4[0] = *(const i4v*)ap;
            a.d4[1] = *(const i4v*)(ap + 16);
            acc[mt] = __builtin_amdgcn_wmma_f32_16x16x32_bf16(
                          false, a.v, false, b.v, (short)0, acc[mt], false, false);
        }
    }
    int col   = nt * 16 + m;
    int mbase = (lane >> 4) << 3;
#pragma unroll
    for (int mt = 0; mt < 4; ++mt)
        for (int v = 0; v < 8; ++v)
            out[(size_t)(mt * 16 + mbase + v) * 1200 + col] = acc[mt][v];
}

// ---------------------------------------------------------------------------
// GRU gate math + length freeze
// ---------------------------------------------------------------------------
__global__ void gru_elem_kernel(const float* __restrict__ gi, const float* __restrict__ gh,
                                const float* __restrict__ bih, const float* __restrict__ bhh,
                                const int* __restrict__ len,
                                float* __restrict__ h, __bf16* __restrict__ h_bf, int t)
{
    int idx = blockIdx.x * blockDim.x + threadIdx.x;
    if (idx >= BB * FF) return;
    int b = idx / FF, f = idx % FF;
    const float* gib = gi + (size_t)b * 1200;
    const float* ghb = gh + (size_t)b * 1200;
    float ir = gib[f]       + bih[f];
    float iz = gib[400 + f] + bih[400 + f];
    float in = gib[800 + f] + bih[800 + f];
    float hr = ghb[f]       + bhh[f];
    float hz = ghb[400 + f] + bhh[400 + f];
    float hn = ghb[800 + f] + bhh[800 + f];
    float r  = 1.0f / (1.0f + expf(-(ir + hr)));
    float z  = 1.0f / (1.0f + expf(-(iz + hz)));
    float nn = tanhf(in + r * hn);
    float hold = h[idx];
    float hnew = (1.0f - z) * nn + z * hold;
    float hv   = (t < len[b]) ? hnew : hold;
    h[idx] = hv;
    h_bf[(size_t)b * HSTRIDE + f] = (__bf16)hv;
}

__global__ void gru_init_kernel(float* __restrict__ h, __bf16* __restrict__ h_bf)
{
    int idx = blockIdx.x * blockDim.x + threadIdx.x;
    if (idx < BB * FF)      h[idx] = 0.0f;
    if (idx < BB * HSTRIDE) h_bf[idx] = (__bf16)0.0f;
}

// ---------------------------------------------------------------------------
// score[b,s] = sum_f new_rep[b,s,f] * hT[b,f]
// ---------------------------------------------------------------------------
__global__ void __launch_bounds__(256)
score_kernel(const float* __restrict__ cand, const float* __restrict__ h,
             float* __restrict__ out)
{
    __shared__ float red[SS * 256];
    int b = blockIdx.x, tid = threadIdx.x;
    float acc[SS] = {0.f, 0.f, 0.f, 0.f, 0.f};
    for (int f = tid; f < FF; f += 256) {
        float hv = h[(size_t)b * FF + f];
        for (int s = 0; s < SS; ++s)
            acc[s] += cand[((size_t)b * SS + s) * FF + f] * hv;
    }
    for (int s = 0; s < SS; ++s) red[s * 256 + tid] = acc[s];
    __syncthreads();
    for (int off = 128; off > 0; off >>= 1) {
        if (tid < off)
            for (int s = 0; s < SS; ++s)
                red[s * 256 + tid] += red[s * 256 + tid + off];
        __syncthreads();
    }
    if (tid < SS) out[b * SS + tid] = red[tid * 256];
}

// ---------------------------------------------------------------------------
extern "C" void kernel_launch(void* const* d_in, const int* in_sizes, int n_in,
                              void* d_out, int out_size, void* d_ws, size_t ws_size,
                              hipStream_t stream)
{
    (void)in_sizes; (void)n_in; (void)out_size; (void)ws_size;
    const float* cand_emb = (const float*)d_in[0];
    const float* clk_emb  = (const float*)d_in[1];
    const int*   clk_len  = (const int*)d_in[2];
    const float* c_conv_w = (const float*)d_in[3];
    const float* c_conv_b = (const float*)d_in[4];
    const float* c_att_W  = (const float*)d_in[5];
    const float* c_att_b  = (const float*)d_in[6];
    const float* c_att_q  = (const float*)d_in[7];
    const float* c_ln_g   = (const float*)d_in[8];
    const float* c_ln_b   = (const float*)d_in[9];
    const float* u_conv_w = (const float*)d_in[10];
    const float* u_conv_b = (const float*)d_in[11];
    const float* u_att_W  = (const float*)d_in[12];
    const float* u_att_b  = (const float*)d_in[13];
    const float* u_att_q  = (const float*)d_in[14];
    const float* u_ln_g   = (const float*)d_in[15];
    const float* u_ln_b   = (const float*)d_in[16];
    const float* gru_wih  = (const float*)d_in[17];
    const float* gru_whh  = (const float*)d_in[18];
    const float* gru_bih  = (const float*)d_in[19];
    const float* gru_bhh  = (const float*)d_in[20];

    uint8_t* ws = (uint8_t*)d_ws;
    size_t off = 0;
    auto alloc = [&](size_t bytes) -> void* {
        void* p = ws + off;
        off = (off + bytes + 255) & ~(size_t)255;
        return p;
    };
    const int convB_elems = CONV_NT * CONV_KC * 512;  // 371200
    const int attB_elems  = ATT_NT  * ATT_KC  * 512;  //  86528
    const int gruB_elems  = GRU_NT  * GRU_KC  * 512;  // 499200

    __bf16* cConvB = (__bf16*)alloc((size_t)convB_elems * 2);
    __bf16* uConvB = (__bf16*)alloc((size_t)convB_elems * 2);
    __bf16* cAttB  = (__bf16*)alloc((size_t)attB_elems * 2);
    __bf16* uAttB  = (__bf16*)alloc((size_t)attB_elems * 2);
    __bf16* wihB   = (__bf16*)alloc((size_t)gruB_elems * 2);
    __bf16* whhB   = (__bf16*)alloc((size_t)gruB_elems * 2);
    float*  cand_rep = (float*)alloc((size_t)BB * SS * FF * 4);
    float*  clk_rep  = (float*)alloc((size_t)BB * HH * FF * 4);
    __bf16* cand_bf  = (__bf16*)alloc((size_t)BB * SS * HSTRIDE * 2);
    __bf16* clk_bf   = (__bf16*)alloc((size_t)BB * HH * HSTRIDE * 2);
    float*  hstate   = (float*)alloc((size_t)BB * FF * 4);
    __bf16* h_bf     = (__bf16*)alloc((size_t)BB * HSTRIDE * 2);
    float*  gi       = (float*)alloc((size_t)BB * 1200 * 4);
    float*  gh       = (float*)alloc((size_t)BB * 1200 * 4);

    // 1) pack weights to bf16 fragment order
    pack_b_kernel<<<(convB_elems + 255) / 256, 256, 0, stream>>>(
        c_conv_w, cConvB, 900, 400, CONV_KC, CONV_NT, 400, 0);
    pack_b_kernel<<<(convB_elems + 255) / 256, 256, 0, stream>>>(
        u_conv_w, uConvB, 900, 400, CONV_KC, CONV_NT, 400, 0);
    pack_b_kernel<<<(attB_elems + 255) / 256, 256, 0, stream>>>(
        c_att_W, cAttB, 400, 200, ATT_KC, ATT_NT, 200, 0);
    pack_b_kernel<<<(attB_elems + 255) / 256, 256, 0, stream>>>(
        u_att_W, uAttB, 400, 200, ATT_KC, ATT_NT, 200, 0);
    pack_b_kernel<<<(gruB_elems + 255) / 256, 256, 0, stream>>>(
        gru_wih, wihB, 400, 1200, GRU_KC, GRU_NT, 400, 1);
    pack_b_kernel<<<(gruB_elems + 255) / 256, 256, 0, stream>>>(
        gru_whh, whhB, 400, 1200, GRU_KC, GRU_NT, 400, 1);

    // 2) encoders (one block per title)
    encoder_kernel<<<BB * SS, 256, 0, stream>>>(
        cand_emb, cConvB, c_conv_b, cAttB, c_att_b, c_att_q, c_ln_g, c_ln_b,
        cand_rep, cand_bf);
    encoder_kernel<<<BB * HH, 256, 0, stream>>>(
        clk_emb, uConvB, u_conv_b, uAttB, u_att_b, u_att_q, u_ln_g, u_ln_b,
        clk_rep, clk_bf);

    // 3) GRU
    gru_init_kernel<<<(BB * HSTRIDE + 255) / 256, 256, 0, stream>>>(hstate, h_bf);
    const int gemm_blocks = (GRU_NT + 7) / 8; // 10
    for (int t = 0; t < HH; ++t) {
        gru_gemm_kernel<<<dim3(gemm_blocks, 2), 256, 0, stream>>>(
            clk_bf, h_bf, wihB, whhB, gi, gh, t);
        gru_elem_kernel<<<(BB * FF + 255) / 256, 256, 0, stream>>>(
            gi, gh, gru_bih, gru_bhh, clk_len, hstate, h_bf, t);
    }

    // 4) scores
    score_kernel<<<BB, 256, 0, stream>>>(cand_rep, hstate, (float*)d_out);
}
